// Transformer_CVAE_36532991820386
// MI455X (gfx1250) — compile-verified, matrix operand order checked by reading, not run on hardware
//
#include <hip/hip_runtime.h>
#include <hip/hip_bf16.h>

// ---------------------------------------------------------------------------
// Transformer encoder layer (post-LN) for MI455X / gfx1250, wave32 + WMMA f16.
// S=1024, B=8, D=512, H=8, HD=64, DFF=2048.  GEMMs via v_wmma_f32_16x16x32_f16
// with double-buffered async (ASYNCcnt) A-tile staging and TDM (TENSORcnt)
// B-tile staging overlapping the WMMA pipeline.
// ---------------------------------------------------------------------------

#define S_LEN 1024
#define B_SZ  8
#define D_MOD 512
#define NHEAD 8
#define HDIM  64
#define DFF   2048
#define NTOK  (S_LEN * B_SZ)   // 8192

typedef _Float16 v16h __attribute__((ext_vector_type(16)));
typedef _Float16 v8h  __attribute__((ext_vector_type(8)));
typedef float    v8f  __attribute__((ext_vector_type(8)));
typedef unsigned int v4u __attribute__((ext_vector_type(4)));
typedef int      v8i  __attribute__((ext_vector_type(8)));
typedef int      v4i  __attribute__((ext_vector_type(4)));

union V16U { v16h v; v8h h[2]; };

__device__ __forceinline__ v8f v8f_zero() {
    v8f z;
#pragma unroll
    for (int i = 0; i < 8; ++i) z[i] = 0.0f;
    return z;
}

__device__ __forceinline__ v16h ld_frag(const _Float16* p) {
    // A/B operand fragment for 16x16x32 f16 WMMA:
    // lane l holds row (l&15); halves 0..7 = K kb*8..+7, halves 8..15 = +16
    V16U u;
    u.h[0] = *(const v8h*)(p);
    u.h[1] = *(const v8h*)(p + 16);
    return u.v;
}

__device__ __forceinline__ v8f wmma32(v16h a, v16h b, v8f c) {
    return __builtin_amdgcn_wmma_f32_16x16x32_f16(
        /*neg_a=*/false, a, /*neg_b=*/false, b,
        /*c_mod=*/(short)0, c, /*reuse_a=*/false, /*reuse_b=*/false);
}

// ---------------------------------------------------------------------------
// f32 -> f16 cast (grid-stride)
// ---------------------------------------------------------------------------
__global__ void cast_f32_f16(const float* __restrict__ in,
                             _Float16* __restrict__ out, int n) {
    for (int i = blockIdx.x * blockDim.x + threadIdx.x; i < n;
         i += gridDim.x * blockDim.x)
        out[i] = (_Float16)in[i];
}

// ---------------------------------------------------------------------------
// Shared GEMM main loop: C[128x64] block tile, 8 waves of 32 threads,
// each wave computes a 32x32 sub-tile as 2x2 WMMA fragments.
//   y[m][n] = sum_k A[m][k] * W[n][k]       (A: [M,K] rows, W: [N,K] rows)
//
// Double-buffered staging: while tile kt is consumed by WMMAs, tile kt+1 is
// in flight (A: per-lane GLOBAL_LOAD_ASYNC_TO_LDS_B128; B: wave-0 TDM
// TENSOR_LOAD_TO_LDS with pad_interval/pad_amount producing the 40-half
// padded rows).  Waits use the counter semantics: async loads complete in
// order, so s_wait_asynccnt 2 / s_wait_tensorcnt 1 release the previous
// tile while the next stays in flight.  Dynamic LDS only => extern
// __shared__ base is LDS offset 0, so LDS addresses are element offsets * 2.
// ---------------------------------------------------------------------------
#define BM 128
#define BN 64
#define LDS_STRIDE 40   // 32 halfs + 8 pad (80B rows, 16B aligned sub-chunks)
#define GEMM_LDS_BYTES (2 * (BM + BN) * LDS_STRIDE * 2)

__device__ __forceinline__ void gemm_mainloop(
    const _Float16* __restrict__ A, const _Float16* __restrict__ W,
    int K, int Nrows, int bm, int bn, _Float16* smem, v8f C[2][2]) {
    const int tid   = threadIdx.x;
    const int w     = tid >> 5;
    const int l     = tid & 31;
    const int wm    = (w & 3) * 32;
    const int wn    = (w >> 2) * 32;
    const int row16 = l & 15;
    const int kb8   = (l >> 4) * 8;

    const int arow = tid >> 1, acol = (tid & 1) * 16;   // A: 16 halfs / thread
    const unsigned ldsA_th = (unsigned)((arow * LDS_STRIDE + acol) * 2);
    const unsigned aBufOff = (unsigned)(BM * LDS_STRIDE * 2);
    const unsigned bBase   = 2u * aBufOff;
    const unsigned bBufOff = (unsigned)(BN * LDS_STRIDE * 2);

    // TDM descriptor group1 (loop-invariant): 2D tile 64 rows x 32 halfs,
    // data_size=2B, pad_enable, pad_interval=16 DWORDs, pad_amount=4 DWORDs.
    v8i g1;
    g1[0] = (int)((1u << 16) | (1u << 20) | (3u << 22) | (3u << 25));
    g1[1] = (int)(((unsigned)K & 0xFFFFu) << 16);               // tensor_dim0 lo
    g1[2] = (int)(((unsigned)K >> 16) |
                  (((unsigned)Nrows & 0xFFFFu) << 16));         // dim0 hi | dim1 lo
    g1[3] = (int)(((unsigned)Nrows >> 16) | (32u << 16));       // dim1 hi | tile_dim0
    g1[4] = 64;                                                 // tile_dim1 (rows)
    g1[5] = (int)K;                                             // dim0_stride lo
    g1[6] = 0;
    g1[7] = 0;
    v4i gz;
    gz[0] = 0; gz[1] = 0; gz[2] = 0; gz[3] = 0;
    v8i gz8;
#pragma unroll
    for (int i = 0; i < 8; ++i) gz8[i] = 0;

    auto issue_tile = [&](int k0, int sel) {
        // A tile: async global -> LDS (per-lane, 2 x 16B)
        const _Float16* pa = A + (size_t)(bm + arow) * K + k0 + acol;
        const unsigned ldsA = ldsA_th + (unsigned)sel * aBufOff;
        asm volatile("global_load_async_to_lds_b128 %0, %1, off"
                     :: "v"(ldsA), "v"(pa) : "memory");
        asm volatile("global_load_async_to_lds_b128 %0, %1, off offset:16"
                     :: "v"(ldsA), "v"(pa) : "memory");
        // B tile: TDM tensor load, wave 0 only
        if (tid < 32) {
            unsigned long long ga =
                (unsigned long long)(uintptr_t)(W + (size_t)bn * K + k0);
            v4u g0;
            g0[0] = 1u;                                        // count=1
            g0[1] = bBase + (unsigned)sel * bBufOff;           // lds_addr
            g0[2] = (unsigned)ga;                              // global_addr lo
            g0[3] = (unsigned)((ga >> 32) & 0x01FFFFFFu) | (2u << 30); // hi|type
            __builtin_amdgcn_tensor_load_to_lds(g0, g1, gz, gz, gz8, 0);
        }
    };

    const int nk = K / 32;
    issue_tile(0, 0);

    for (int kt = 0; kt < nk; ++kt) {
        const int sel = kt & 1;

        if (kt + 1 < nk) {
            issue_tile((kt + 1) * 32, sel ^ 1);
            // release the previous tile only; next tile stays in flight
            asm volatile("s_wait_asynccnt 0x2" ::: "memory");
            if (tid < 32) __builtin_amdgcn_s_wait_tensorcnt(1);
        } else {
            asm volatile("s_wait_asynccnt 0x0" ::: "memory");
            if (tid < 32) __builtin_amdgcn_s_wait_tensorcnt(0);
        }
        if (kt + 2 < nk) {  // warm L2 for the tile after next
            __builtin_prefetch(A + (size_t)(bm + arow) * K + (kt + 2) * 32 + acol,
                               0, 1);
        }
        __syncthreads();    // publish tile kt to all waves

        const _Float16* As = smem + sel * (BM * LDS_STRIDE);
        const _Float16* Bs = smem + 2 * (BM * LDS_STRIDE) + sel * (BN * LDS_STRIDE);

        v16h a0 = ld_frag(&As[(wm + row16) * LDS_STRIDE + kb8]);
        v16h a1 = ld_frag(&As[(wm + 16 + row16) * LDS_STRIDE + kb8]);
        v16h b0 = ld_frag(&Bs[(wn + row16) * LDS_STRIDE + kb8]);
        v16h b1 = ld_frag(&Bs[(wn + 16 + row16) * LDS_STRIDE + kb8]);

        C[0][0] = wmma32(a0, b0, C[0][0]);
        C[0][1] = wmma32(a0, b1, C[0][1]);
        C[1][0] = wmma32(a1, b0, C[1][0]);
        C[1][1] = wmma32(a1, b1, C[1][1]);

        __syncthreads();    // all waves done reading buf sel before reuse
    }
}

// ---------------------------------------------------------------------------
// Generic GEMM: out = act( A @ W^T + bias ) [+ resid], f32 and/or f16 outputs
// ---------------------------------------------------------------------------
__global__ __launch_bounds__(256) void gemm_f16(
    const _Float16* __restrict__ A, const _Float16* __restrict__ W,
    const float* __restrict__ bias, const float* __restrict__ resid,
    float* __restrict__ outF, _Float16* __restrict__ outH,
    int M, int N, int K, int relu) {
    extern __shared__ _Float16 dsm[];

    const int nblk = N / BN;
    const int bm = (blockIdx.x / nblk) * BM;
    const int bn = (blockIdx.x % nblk) * BN;

    v8f C[2][2];
#pragma unroll
    for (int i = 0; i < 2; ++i)
#pragma unroll
        for (int j = 0; j < 2; ++j) C[i][j] = v8f_zero();

    gemm_mainloop(A, W, K, N, bm, bn, dsm, C);

    const int l     = threadIdx.x & 31;
    const int w     = threadIdx.x >> 5;
    const int wm    = (w & 3) * 32;
    const int wn    = (w >> 2) * 32;
    const int row16 = l & 15;
    const int mlo   = (l >> 4) * 8;   // row offset for C layout

#pragma unroll
    for (int i = 0; i < 2; ++i)
#pragma unroll
        for (int j = 0; j < 2; ++j) {
            const int ng = bn + wn + j * 16 + row16;
            const float bv = bias ? bias[ng] : 0.0f;
#pragma unroll
            for (int r = 0; r < 8; ++r) {
                const int mg = bm + wm + i * 16 + mlo + r;
                float v = C[i][j][r] + bv;
                if (relu) v = fmaxf(v, 0.0f);
                if (resid) v += resid[(size_t)mg * N + ng];
                if (outF) outF[(size_t)mg * N + ng] = v;
                if (outH) outH[(size_t)mg * N + ng] = (_Float16)v;
            }
        }
}

// ---------------------------------------------------------------------------
// QKV GEMM with scatter epilogue into attention layouts:
//   q,k : [B*H, S, HD]   (rows contiguous over HD)
//   vT  : [B*H, HD, S]   (rows contiguous over S  -> PV B-operand contiguous)
// ---------------------------------------------------------------------------
__global__ __launch_bounds__(256) void gemm_qkv(
    const _Float16* __restrict__ xh, const _Float16* __restrict__ wqkv,
    const float* __restrict__ bias,
    _Float16* __restrict__ qh, _Float16* __restrict__ kh,
    _Float16* __restrict__ vth) {
    extern __shared__ _Float16 dsm[];

    const int N = 3 * D_MOD;
    const int nblk = N / BN;
    const int bm = (blockIdx.x / nblk) * BM;
    const int bn = (blockIdx.x % nblk) * BN;

    v8f C[2][2];
#pragma unroll
    for (int i = 0; i < 2; ++i)
#pragma unroll
        for (int j = 0; j < 2; ++j) C[i][j] = v8f_zero();

    gemm_mainloop(xh, wqkv, D_MOD, N, bm, bn, dsm, C);

    const int l     = threadIdx.x & 31;
    const int w     = threadIdx.x >> 5;
    const int wm    = (w & 3) * 32;
    const int wn    = (w >> 2) * 32;
    const int row16 = l & 15;
    const int mlo   = (l >> 4) * 8;

#pragma unroll
    for (int i = 0; i < 2; ++i)
#pragma unroll
        for (int j = 0; j < 2; ++j) {
            const int ng = bn + wn + j * 16 + row16;   // [0, 1536)
            const int reg = ng >> 9;                   // 0=q 1=k 2=v
            const int c   = ng & 511;
            const int h   = c >> 6;
            const int hd  = c & 63;
            const float bv = bias[ng];
#pragma unroll
            for (int r = 0; r < 8; ++r) {
                const int mg = bm + wm + i * 16 + mlo + r;  // token = s*B + b
                const int s  = mg >> 3;
                const int b  = mg & 7;
                const int bh = b * NHEAD + h;
                const _Float16 hv = (_Float16)(C[i][j][r] + bv);
                if (reg == 0)
                    qh[((size_t)bh * S_LEN + s) * HDIM + hd] = hv;
                else if (reg == 1)
                    kh[((size_t)bh * S_LEN + s) * HDIM + hd] = hv;
                else
                    vth[((size_t)bh * HDIM + hd) * S_LEN + s] = hv;
            }
        }
}

// ---------------------------------------------------------------------------
// T5-style relative-position bias (bidirectional, 32 buckets, max dist 128)
// ---------------------------------------------------------------------------
__device__ __forceinline__ float rel_bias_val(const float* __restrict__ rel,
                                              int h, int d) {
    const int rb = (d > 0) ? 16 : 0;
    const int rp = (d < 0) ? -d : d;
    int bucket;
    if (rp < 8) {
        bucket = rb + rp;
    } else {
        // 8 + log(rp/8)/log(16)*8 , clamped to 15
        float t = __logf((float)rp * 0.125f) * (8.0f / 2.7725887f);
        int lg = 8 + (int)t;
        bucket = rb + (lg < 15 ? lg : 15);
    }
    return rel[bucket * NHEAD + h];
}

// ---------------------------------------------------------------------------
// Flash attention: each wave owns 16 query rows; loops over keys in chunks
// of 32 with online softmax.  grid = (S/128) * (B*H), block = 256 (8 waves).
// ---------------------------------------------------------------------------
__global__ __launch_bounds__(256) void attn_kernel(
    const _Float16* __restrict__ qh, const _Float16* __restrict__ kh,
    const _Float16* __restrict__ vth, const float* __restrict__ mask,
    const float* __restrict__ rel, _Float16* __restrict__ oh) {
    __shared__ _Float16 P[8][16 * 32];   // per-wave P tile (C-layout -> A-layout)

    const int bh = blockIdx.x & (B_SZ * NHEAD - 1);
    const int qc = blockIdx.x >> 6;
    const int b  = bh >> 3;
    const int h  = bh & 7;

    const int w     = threadIdx.x >> 5;
    const int l     = threadIdx.x & 31;
    const int row16 = l & 15;
    const int kb    = l >> 4;
    const int kb8   = kb * 8;
    const int m0    = qc * 128 + w * 16;

    _Float16* Pw = P[w];

    // Q fragments (16 rows x 64 K, two 16x32 fragments)
    const _Float16* qbase = qh + ((size_t)bh * S_LEN + m0 + row16) * HDIM;
    const v16h aq0 = ld_frag(qbase + kb8);
    const v16h aq1 = ld_frag(qbase + 32 + kb8);

    float mi[8], li[8];
    v8f oa[4];
#pragma unroll
    for (int r = 0; r < 8; ++r) { mi[r] = -1e30f; li[r] = 0.0f; }
#pragma unroll
    for (int j = 0; j < 4; ++j) oa[j] = v8f_zero();

    for (int kc = 0; kc < S_LEN / 32; ++kc) {
        const int j0 = kc * 32;

        // scores: two 16x16 C frags covering 32 key columns
        const _Float16* kb0 = kh + ((size_t)bh * S_LEN + j0 + row16) * HDIM;
        const _Float16* kb1 = kb0 + 16 * HDIM;
        v8f c0 = wmma32(aq0, ld_frag(kb0 + kb8), v8f_zero());
        c0     = wmma32(aq1, ld_frag(kb0 + 32 + kb8), c0);
        v8f c1 = wmma32(aq0, ld_frag(kb1 + kb8), v8f_zero());
        c1     = wmma32(aq1, ld_frag(kb1 + 32 + kb8), c1);

        float scl[8];
        const int c0g = j0 + row16;
        const int c1g = c0g + 16;
#pragma unroll
        for (int r = 0; r < 8; ++r) {
            const int mg = m0 + r + 8 * kb;
            float v0 = c0[r] * 0.125f + mask[(size_t)mg * S_LEN + c0g] +
                       rel_bias_val(rel, h, c0g - mg);
            float v1 = c1[r] * 0.125f + mask[(size_t)mg * S_LEN + c1g] +
                       rel_bias_val(rel, h, c1g - mg);
            // row max over 16 lanes (row of C lives across a 16-lane group)
            float mx = fmaxf(v0, v1);
#pragma unroll
            for (int off = 8; off >= 1; off >>= 1)
                mx = fmaxf(mx, __shfl_xor(mx, off, 16));
            const float nm = fmaxf(mi[r], mx);
            const float sc = __expf(mi[r] - nm);
            const float p0 = __expf(v0 - nm);
            const float p1 = __expf(v1 - nm);
            float sum = p0 + p1;
#pragma unroll
            for (int off = 8; off >= 1; off >>= 1)
                sum += __shfl_xor(sum, off, 16);
            li[r] = li[r] * sc + sum;
            mi[r] = nm;
            scl[r] = sc;
            // write P in row-major (16 x 32) for A-fragment reload
            Pw[(r + 8 * kb) * 32 + row16]      = (_Float16)p0;
            Pw[(r + 8 * kb) * 32 + row16 + 16] = (_Float16)p1;
        }
#pragma unroll
        for (int j = 0; j < 4; ++j)
#pragma unroll
            for (int r = 0; r < 8; ++r) oa[j][r] *= scl[r];

        __syncthreads();
        const v16h ap = ld_frag(&Pw[row16 * 32 + kb8]);   // P as A operand
#pragma unroll
        for (int j = 0; j < 4; ++j) {
            const int hd = j * 16 + row16;
            const _Float16* vb =
                vth + ((size_t)bh * HDIM + hd) * S_LEN + j0 + kb8;
            oa[j] = wmma32(ap, ld_frag(vb), oa[j]);
        }
        __syncthreads();
    }

    // normalize + scatter to [token, D] row-major f16 for out-proj GEMM
#pragma unroll
    for (int r = 0; r < 8; ++r) {
        const float inv = 1.0f / li[r];
        const int s = m0 + r + 8 * kb;
        const size_t orow = (size_t)(s * B_SZ + b) * D_MOD + h * HDIM;
#pragma unroll
        for (int j = 0; j < 4; ++j)
            oh[orow + j * 16 + row16] = (_Float16)(oa[j][r] * inv);
    }
}

// ---------------------------------------------------------------------------
// LayerNorm over D=512: one row per block, 256 threads (2 elems each)
// ---------------------------------------------------------------------------
__global__ __launch_bounds__(256) void ln_kernel(
    const float* __restrict__ in, const float* __restrict__ wt,
    const float* __restrict__ bs, float* __restrict__ outF,
    _Float16* __restrict__ outH) {
    __shared__ float red[8];
    const int row = blockIdx.x;
    const int t = threadIdx.x;
    const float* p = in + (size_t)row * D_MOD;
    const float a = p[t];
    const float c = p[t + 256];

    float s = a + c;
#pragma unroll
    for (int off = 16; off >= 1; off >>= 1) s += __shfl_xor(s, off, 32);
    if ((t & 31) == 0) red[t >> 5] = s;
    __syncthreads();
    float tot = 0.0f;
#pragma unroll
    for (int i = 0; i < 8; ++i) tot += red[i];
    const float mu = tot * (1.0f / (float)D_MOD);
    __syncthreads();

    const float d0 = a - mu, d1 = c - mu;
    float q = d0 * d0 + d1 * d1;
#pragma unroll
    for (int off = 16; off >= 1; off >>= 1) q += __shfl_xor(q, off, 32);
    if ((t & 31) == 0) red[t >> 5] = q;
    __syncthreads();
    float vtot = 0.0f;
#pragma unroll
    for (int i = 0; i < 8; ++i) vtot += red[i];
    const float rstd = rsqrtf(vtot * (1.0f / (float)D_MOD) + 1e-5f);

    const float o0 = d0 * rstd * wt[t] + bs[t];
    const float o1 = d1 * rstd * wt[t + 256] + bs[t + 256];
    if (outF) {
        outF[(size_t)row * D_MOD + t] = o0;
        outF[(size_t)row * D_MOD + t + 256] = o1;
    }
    if (outH) {
        outH[(size_t)row * D_MOD + t] = (_Float16)o0;
        outH[(size_t)row * D_MOD + t + 256] = (_Float16)o1;
    }
}

// ---------------------------------------------------------------------------
// Host launcher
// ---------------------------------------------------------------------------
extern "C" void kernel_launch(void* const* d_in, const int* in_sizes, int n_in,
                              void* d_out, int out_size, void* d_ws,
                              size_t ws_size, hipStream_t stream) {
    (void)in_sizes; (void)n_in; (void)out_size; (void)ws_size;

    const float* x          = (const float*)d_in[0];   // [S,B,D] = [8192,512]
    const float* attn_mask  = (const float*)d_in[1];   // [S,S]
    const float* in_proj_w  = (const float*)d_in[2];   // [1536,512]
    const float* in_proj_b  = (const float*)d_in[3];
    const float* out_proj_w = (const float*)d_in[4];   // [512,512]
    const float* out_proj_b = (const float*)d_in[5];
    const float* lin1_w     = (const float*)d_in[6];   // [2048,512]
    const float* lin1_b     = (const float*)d_in[7];
    const float* lin2_w     = (const float*)d_in[8];   // [512,2048]
    const float* lin2_b     = (const float*)d_in[9];
    const float* ln1_w      = (const float*)d_in[10];
    const float* ln1_b      = (const float*)d_in[11];
    const float* ln2_w      = (const float*)d_in[12];
    const float* ln2_b      = (const float*)d_in[13];
    const float* rel_emb    = (const float*)d_in[14];  // [32,8]

    uint8_t* ws = (uint8_t*)d_ws;
    size_t off = 0;
    auto carve = [&](size_t bytes) -> void* {
        void* p = ws + off;
        off += (bytes + 255) & ~(size_t)255;
        return p;
    };

    const size_t ND   = (size_t)NTOK * D_MOD;        // 4,194,304
    const size_t NDFF = (size_t)NTOK * DFF;          // 16,777,216

    _Float16* xh    = (_Float16*)carve(ND * 2);
    _Float16* wqkvh = (_Float16*)carve((size_t)3 * D_MOD * D_MOD * 2);
    _Float16* woh   = (_Float16*)carve((size_t)D_MOD * D_MOD * 2);
    _Float16* w1h   = (_Float16*)carve((size_t)DFF * D_MOD * 2);
    _Float16* w2h   = (_Float16*)carve((size_t)D_MOD * DFF * 2);
    _Float16* qh    = (_Float16*)carve(ND * 2);
    _Float16* kh    = (_Float16*)carve(ND * 2);
    _Float16* vth   = (_Float16*)carve(ND * 2);
    _Float16* oh    = (_Float16*)carve(ND * 2);
    _Float16* ff1h  = (_Float16*)carve(NDFF * 2);
    float*    res1  = (float*)carve(ND * 4);
    float*    yf    = (float*)carve(ND * 4);
    _Float16* yh    = xh;     // xh dead after QKV GEMM
    float*    ff2   = res1;   // res1 dead after LN1

    // 1) casts to f16
    cast_f32_f16<<<1024, 256, 0, stream>>>(x, xh, (int)ND);
    cast_f32_f16<<<512, 256, 0, stream>>>(in_proj_w, wqkvh, 3 * D_MOD * D_MOD);
    cast_f32_f16<<<256, 256, 0, stream>>>(out_proj_w, woh, D_MOD * D_MOD);
    cast_f32_f16<<<512, 256, 0, stream>>>(lin1_w, w1h, DFF * D_MOD);
    cast_f32_f16<<<512, 256, 0, stream>>>(lin2_w, w2h, D_MOD * DFF);

    // 2) QKV projection + scatter into attention layouts
    gemm_qkv<<<(NTOK / BM) * (3 * D_MOD / BN), 256, GEMM_LDS_BYTES, stream>>>(
        xh, wqkvh, in_proj_b, qh, kh, vth);

    // 3) flash attention with relative-position bias
    attn_kernel<<<(S_LEN / 128) * (B_SZ * NHEAD), 256, 0, stream>>>(
        qh, kh, vth, attn_mask, rel_emb, oh);

    // 4) out-proj + residual(x)  -> res1 (f32)
    gemm_f16<<<(NTOK / BM) * (D_MOD / BN), 256, GEMM_LDS_BYTES, stream>>>(
        oh, woh, out_proj_b, x, res1, nullptr, NTOK, D_MOD, D_MOD, 0);

    // 5) LN1 -> yf (f32 residual) + yh (f16 GEMM input)
    ln_kernel<<<NTOK, 256, 0, stream>>>(res1, ln1_w, ln1_b, yf, yh);

    // 6) FFN up + ReLU -> ff1h (f16)
    gemm_f16<<<(NTOK / BM) * (DFF / BN), 256, GEMM_LDS_BYTES, stream>>>(
        yh, w1h, lin1_b, nullptr, nullptr, ff1h, NTOK, DFF, D_MOD, 1);

    // 7) FFN down + residual(y) -> ff2 (f32)
    gemm_f16<<<(NTOK / BM) * (D_MOD / BN), 256, GEMM_LDS_BYTES, stream>>>(
        ff1h, w2h, lin2_b, yf, ff2, nullptr, NTOK, D_MOD, DFF, 0);

    // 8) LN2 -> d_out (f32)
    ln_kernel<<<NTOK, 256, 0, stream>>>(ff2, ln2_w, ln2_b, (float*)d_out,
                                        nullptr);
}
